// Grouped_Query_Attention_57226144252304
// MI455X (gfx1250) — compile-verified, matrix operand order checked by reading, not run on hardware
//
#include <hip/hip_runtime.h>
#include <hip/hip_bf16.h>

#define USE_TDM 1   // flip to 0 if tensor_load_to_lds asm fails to assemble

typedef __attribute__((ext_vector_type(16))) __bf16 v16bf;
typedef __attribute__((ext_vector_type(8)))  __bf16 v8bf;
typedef __attribute__((ext_vector_type(4)))  __bf16 v4bf;
typedef __attribute__((ext_vector_type(8)))  float  v8f;
typedef __attribute__((ext_vector_type(4)))  unsigned int u32x4;
typedef __attribute__((ext_vector_type(8)))  unsigned int u32x8;

#define B_      16
#define S_      4
#define D_      4096
#define NH_     32
#define NKV_    8
#define HD_     128
#define REP_    4
#define MAXSEQ_ 8192
#define START_  8188
#define T_      (START_ + S_)          // 8192
#define NEGINF_ (-1.0e9f)
#define NSPLIT  16
#define KEYS_PER_SPLIT (T_ / NSPLIT)   // 512
#define SCALE_  0.08838834764831845f   // 1/sqrt(128)

// native hardware converts (v_cvt_*bf16*) instead of manual RNE bit math
__device__ __forceinline__ __bf16 f2bf(float f) { return (__bf16)f; }

// pack 4 floats -> 4 bf16, 8-byte LDS store
__device__ __forceinline__ void st4bf(__bf16* p, float4 v) {
    v4bf t;
    t[0] = (__bf16)v.x; t[1] = (__bf16)v.y;
    t[2] = (__bf16)v.z; t[3] = (__bf16)v.w;
    *reinterpret_cast<v4bf*>(p) = t;
}

// WMMA 16-bit fragment (ISA 7.12.2): element i of lane l reads
// row[ ((i&8)<<1) + (i&7) + 8*(l>=16) ]  -> two contiguous 8-elem (16B) runs.
__device__ __forceinline__ v16bf ld_frag(const __bf16* rowp, int half) {
    v8bf lo = *reinterpret_cast<const v8bf*>(rowp + half * 8);
    v8bf hi = *reinterpret_cast<const v8bf*>(rowp + 16 + half * 8);
    return __builtin_shufflevector(lo, hi, 0, 1, 2, 3, 4, 5, 6, 7,
                                           8, 9, 10, 11, 12, 13, 14, 15);
}
// same fragment but sourced from fp32 (global Q rows or fp32 LDS K tile)
__device__ __forceinline__ v16bf ld_frag_f32(const float* rowp, int half) {
    float4 a = *reinterpret_cast<const float4*>(rowp + half * 8);
    float4 b = *reinterpret_cast<const float4*>(rowp + half * 8 + 4);
    float4 c = *reinterpret_cast<const float4*>(rowp + 16 + half * 8);
    float4 d = *reinterpret_cast<const float4*>(rowp + 16 + half * 8 + 4);
    v16bf r;
    r[0]  = f2bf(a.x); r[1]  = f2bf(a.y); r[2]  = f2bf(a.z); r[3]  = f2bf(a.w);
    r[4]  = f2bf(b.x); r[5]  = f2bf(b.y); r[6]  = f2bf(b.z); r[7]  = f2bf(b.w);
    r[8]  = f2bf(c.x); r[9]  = f2bf(c.y); r[10] = f2bf(c.z); r[11] = f2bf(c.w);
    r[12] = f2bf(d.x); r[13] = f2bf(d.y); r[14] = f2bf(d.z); r[15] = f2bf(d.w);
    return r;
}

// stage one 32(k) x 16(n) fp32 W tile transposed into LDS [n][k] (bf16):
// lane: n-group = l>>3 (4 cols), 4 consecutive k-rows at 4*(l&7) -> b64 stores
__device__ __forceinline__ void stage_w_tile(__bf16* wTt, const float* Bm,
                                             int ldb, int k0, int n0, int l) {
    const int cgrp  = l >> 3;           // 0..3
    const int krow4 = (l & 7) * 4;      // 0,4,...,28
    v4bf w0, w1, w2, w3;
    #pragma unroll
    for (int q4 = 0; q4 < 4; ++q4) {
        int r2 = krow4 + q4;
        float4 vb = *reinterpret_cast<const float4*>(Bm + (size_t)(k0 + r2) * ldb + n0 + cgrp * 4);
        w0[q4] = f2bf(vb.x); w1[q4] = f2bf(vb.y);
        w2[q4] = f2bf(vb.z); w3[q4] = f2bf(vb.w);
    }
    *reinterpret_cast<v4bf*>(&wTt[(cgrp * 4 + 0) * 32 + krow4]) = w0;
    *reinterpret_cast<v4bf*>(&wTt[(cgrp * 4 + 1) * 32 + krow4]) = w1;
    *reinterpret_cast<v4bf*>(&wTt[(cgrp * 4 + 2) * 32 + krow4]) = w2;
    *reinterpret_cast<v4bf*>(&wTt[(cgrp * 4 + 3) * 32 + krow4]) = w3;
}

// ---------------------------------------------------------------------------
// Kernel 1: QKV projection with fused RoPE. One wave per 16x16 tile.
// ---------------------------------------------------------------------------
__global__ __launch_bounds__(32)
void qkv_rope_kernel(const float* __restrict__ x,
                     const float* __restrict__ wq,
                     const float* __restrict__ wk,
                     const float* __restrict__ wv,
                     const float* __restrict__ fcos,
                     const float* __restrict__ fsin,
                     float* __restrict__ q_ws,
                     float* __restrict__ k_ws,
                     float* __restrict__ v_ws) {
    __shared__ __bf16 aT[16 * 32];    // [m][k]
    __shared__ __bf16 wTt[16 * 32];   // [n][k]  (transposed W tile)
    const int l    = threadIdx.x;
    const int half = l >> 4, mn = l & 15;
    const int mt = blockIdx.x;
    const int nt = blockIdx.y;

    const float* Bm; int ldb, n0, region;
    if (nt < 256)      { Bm = wq; ldb = NH_  * HD_; n0 = nt * 16;         region = 0; }
    else if (nt < 320) { Bm = wk; ldb = NKV_ * HD_; n0 = (nt - 256) * 16; region = 1; }
    else               { Bm = wv; ldb = NKV_ * HD_; n0 = (nt - 320) * 16; region = 2; }
    const int m0 = mt * 16;

    v8f acc = {};
    #pragma unroll 1
    for (int k0 = 0; k0 < D_; k0 += 32) {
        #pragma unroll
        for (int q4 = 0; q4 < 4; ++q4) {
            int idx = l + 32 * q4;                    // A: 16x32 row-major
            int r  = idx >> 3, c4 = idx & 7;
            float4 va = *reinterpret_cast<const float4*>(x + (size_t)(m0 + r) * D_ + k0 + c4 * 4);
            st4bf(&aT[r * 32 + c4 * 4], va);
        }
        stage_w_tile(wTt, Bm, ldb, k0, n0, l);
        __syncthreads();
        v16bf a  = ld_frag(&aT[mn * 32], half);
        v16bf bf = ld_frag(&wTt[mn * 32], half);
        acc = __builtin_amdgcn_wmma_f32_16x16x32_bf16(false, a, false, bf, (short)0, acc, false, false);
        __syncthreads();
    }

    // C layout: lane holds column n = l&15, rows m = vg + 8*(l>>4)
    if (region == 2) {
        #pragma unroll
        for (int vg = 0; vg < 8; ++vg) {
            int R    = m0 + vg + 8 * half;
            int bidx = R >> 2, spos = R & 3;
            int ncol = n0 + mn;
            int h = ncol >> 7, d = ncol & 127;
            v_ws[((size_t)(bidx * NKV_ + h) * S_ + spos) * HD_ + d] = acc[vg];
        }
    } else {
        float* dst = (region == 0) ? q_ws : k_ws;
        int    nh  = (region == 0) ? NH_  : NKV_;
        #pragma unroll
        for (int vg = 0; vg < 8; ++vg) {
            int R    = m0 + vg + 8 * half;
            int bidx = R >> 2, spos = R & 3;
            int ncol = n0 + mn;
            int dh   = ncol & (HD_ - 1);
            int pair = dh >> 1;
            float val   = acc[vg];
            float other = __shfl_xor(val, 1, 32);   // RoPE pair partner (col n^1)
            float c = fcos[spos * (HD_ / 2) + pair];
            float s = fsin[spos * (HD_ / 2) + pair];
            val = (dh & 1) ? (other * s + val * c) : (val * c - other * s);
            int h = ncol >> 7;
            dst[((size_t)(bidx * nh + h) * S_ + spos) * HD_ + dh] = val;
        }
    }
}

// ---------------------------------------------------------------------------
// Kernel 2: flash-decoding attention. One wave per (split, kv, b).
// K tile streamed by the Tensor Data Mover into fp32 LDS (fast path);
// V tile staged via VGPRs with bf16 transpose; softmax row-sum via ones-WMMA.
// ---------------------------------------------------------------------------
__global__ __launch_bounds__(32)
void attn_kernel(const float* __restrict__ cache_k,
                 const float* __restrict__ cache_v,
                 const float* __restrict__ q_ws,
                 const float* __restrict__ k_ws,
                 const float* __restrict__ v_ws,
                 float* __restrict__ acc_ws,
                 float* __restrict__ ml_ws) {
    __shared__ float  kTf[32 * HD_];   // [key][d] fp32 (TDM destination)
    __shared__ __bf16 vTt[HD_ * 32];   // [d][key] bf16
    __shared__ __bf16 pT[16 * 32];     // [row][key] bf16
    const int l     = threadIdx.x;
    const int half  = l >> 4, mn = l & 15;
    const int split = blockIdx.x;
    const int kv    = blockIdx.y;
    const int b     = blockIdx.z;
    const int d0    = 4 * l;
    const size_t KVSTRIDE = (size_t)NKV_ * HD_;   // 1024 floats between keys

    // Q fragments: rows = rep*S + s, 4 k-chunks of 32 over HD=128
    v16bf qf[4];
    {
        int rep = mn >> 2, spos = mn & 3;
        int h = kv * REP_ + rep;
        const float* qrow = q_ws + ((size_t)(b * NH_ + h) * S_ + spos) * HD_;
        #pragma unroll
        for (int cc = 0; cc < 4; ++cc)
            qf[cc] = ld_frag_f32(qrow + cc * 32, half);
    }
    v16bf ones;
    #pragma unroll
    for (int i = 0; i < 16; ++i) ones[i] = (__bf16)1.0f;

    v8f zero = {};
    v8f o[8];
    #pragma unroll
    for (int c = 0; c < 8; ++c) o[c] = zero;
    v8f osum = zero;                    // row sums (flash-attn "l"), via WMMA
    float mi[8];
    #pragma unroll
    for (int vg = 0; vg < 8; ++vg) mi[vg] = -1.0e30f;

    const int jbase0 = split * KEYS_PER_SPLIT;
    #pragma unroll 1
    for (int blk = 0; blk < KEYS_PER_SPLIT; blk += 32) {
        const int jbase = jbase0 + blk;

        // prefetch next V block (one 512B row per lane) -> global_prefetch_b8
        if (blk + 32 < KEYS_PER_SPLIT) {
            int jn = jbase + 32 + l;
            if (jn < START_) {
                size_t off = ((size_t)(b * MAXSEQ_ + jn) * NKV_ + kv) * HD_;
                __builtin_prefetch(cache_v + off, 0, 0);
            }
        }

        if (jbase + 32 <= START_) {
            // ---- fast path (all but the final boundary block)
            const size_t coff = ((size_t)(b * MAXSEQ_ + jbase) * NKV_ + kv) * HD_;
#if USE_TDM
            {
                // TDM D#: 2D tile, 128 x 32 fp32 elements, row stride 1024 elts
                unsigned long long gaddr = (unsigned long long)(const void*)(cache_k + coff);
                unsigned ldsaddr = (unsigned)(size_t)(void*)kTf;
                u32x4 g0;
                g0[0] = 1u;                                        // count=1
                g0[1] = ldsaddr;                                   // lds_addr
                g0[2] = (unsigned)(gaddr & 0xFFFFFFFFu);           // global_addr lo
                g0[3] = ((unsigned)(gaddr >> 32) & 0x01FFFFFFu)    // global_addr hi
                        | (2u << 30);                              // type=2 (image)
                u32x8 g1;
                g1[0] = 2u << 16;                                  // data_size=4B
                g1[1] = 128u << 16;                                // tensor_dim0=128 (lo16)
                g1[2] = 0u;                                        // dim0 hi / dim1 lo
                g1[3] = (128u << 16) | 0x10u;                      // tile_dim0=128 | dim1 hi (dim1=1<<20)
                g1[4] = 32u;                                       // tile_dim1=32
                g1[5] = 1024u;                                     // tensor_dim0_stride
                g1[6] = 0u;
                g1[7] = 0u;
                asm volatile("tensor_load_to_lds %0, %1" :: "s"(g0), "s"(g1) : "memory");
            }
#else
            const float* kb = cache_k + coff + d0;
#endif
            const float* vb = cache_v + coff + d0;
            #pragma unroll 1
            for (int g = 0; g < 4; ++g) {
                v8bf t0, t1, t2, t3;
                #pragma unroll
                for (int u = 0; u < 8; ++u) {
                    int t = g * 8 + u;
#if !USE_TDM
                    float4 ka = *reinterpret_cast<const float4*>(kb + (size_t)t * KVSTRIDE);
                    *reinterpret_cast<float4*>(&kTf[t * HD_ + d0]) = ka;
#endif
                    float4 va = *reinterpret_cast<const float4*>(vb + (size_t)t * KVSTRIDE);
                    t0[u] = f2bf(va.x); t1[u] = f2bf(va.y);
                    t2[u] = f2bf(va.z); t3[u] = f2bf(va.w);
                }
                *reinterpret_cast<v8bf*>(&vTt[(d0 + 0) * 32 + g * 8]) = t0;
                *reinterpret_cast<v8bf*>(&vTt[(d0 + 1) * 32 + g * 8]) = t1;
                *reinterpret_cast<v8bf*>(&vTt[(d0 + 2) * 32 + g * 8]) = t2;
                *reinterpret_cast<v8bf*>(&vTt[(d0 + 3) * 32 + g * 8]) = t3;
            }
#if USE_TDM
            __builtin_amdgcn_s_wait_tensorcnt(0);
#endif
        } else {
            // ---- boundary block: last 4 keys come from freshly projected K/V
            #pragma unroll 1
            for (int g = 0; g < 4; ++g) {
                v8bf t0, t1, t2, t3;
                #pragma unroll
                for (int u = 0; u < 8; ++u) {
                    int t = g * 8 + u;
                    int j = jbase + t;                  // uniform across lanes
                    const float *kp, *vp;
                    if (j >= START_) {
                        size_t off = ((size_t)(b * NKV_ + kv) * S_ + (j - START_)) * HD_ + d0;
                        kp = k_ws + off; vp = v_ws + off;
                    } else {
                        size_t off = ((size_t)(b * MAXSEQ_ + j) * NKV_ + kv) * HD_ + d0;
                        kp = cache_k + off; vp = cache_v + off;
                    }
                    float4 ka = *reinterpret_cast<const float4*>(kp);
                    float4 va = *reinterpret_cast<const float4*>(vp);
                    *reinterpret_cast<float4*>(&kTf[t * HD_ + d0]) = ka;
                    t0[u] = f2bf(va.x); t1[u] = f2bf(va.y);
                    t2[u] = f2bf(va.z); t3[u] = f2bf(va.w);
                }
                *reinterpret_cast<v8bf*>(&vTt[(d0 + 0) * 32 + g * 8]) = t0;
                *reinterpret_cast<v8bf*>(&vTt[(d0 + 1) * 32 + g * 8]) = t1;
                *reinterpret_cast<v8bf*>(&vTt[(d0 + 2) * 32 + g * 8]) = t2;
                *reinterpret_cast<v8bf*>(&vTt[(d0 + 3) * 32 + g * 8]) = t3;
            }
        }
        __syncthreads();

        // ---- scores: two 16x16 tiles over 32 keys, K-dim = HD in 4 chunks
        v8f s0 = zero, s1 = zero;
        #pragma unroll
        for (int cc = 0; cc < 4; ++cc) {
            v16bf b0 = ld_frag_f32(&kTf[mn * HD_ + cc * 32], half);
            v16bf b1 = ld_frag_f32(&kTf[(16 + mn) * HD_ + cc * 32], half);
            s0 = __builtin_amdgcn_wmma_f32_16x16x32_bf16(false, qf[cc], false, b0, (short)0, s0, false, false);
            s1 = __builtin_amdgcn_wmma_f32_16x16x32_bf16(false, qf[cc], false, b1, (short)0, s1, false, false);
        }

        // ---- scale + (rare) causal mask + online softmax
        const bool need_mask = (jbase + 32 > START_);   // uniform
        int j0 = jbase + mn, j1 = jbase + 16 + mn;
        #pragma unroll
        for (int vg = 0; vg < 8; ++vg) {
            int m = vg + 8 * half;
            float a0 = s0[vg] * SCALE_;
            float a1 = s1[vg] * SCALE_;
            if (need_mask) {
                int spos = m & 3;
                if (j0 > START_ + spos) a0 = NEGINF_;
                if (j1 > START_ + spos) a1 = NEGINF_;
            }
            float tmax = fmaxf(a0, a1);
            tmax = fmaxf(tmax, __shfl_xor(tmax, 1, 32));
            tmax = fmaxf(tmax, __shfl_xor(tmax, 2, 32));
            tmax = fmaxf(tmax, __shfl_xor(tmax, 4, 32));
            tmax = fmaxf(tmax, __shfl_xor(tmax, 8, 32));
            float mnew = fmaxf(mi[vg], tmax);
            float f  = __expf(mi[vg] - mnew);
            float p0 = __expf(a0 - mnew);
            float p1 = __expf(a1 - mnew);
            mi[vg] = mnew;
            osum[vg] *= f;
            #pragma unroll
            for (int c = 0; c < 8; ++c) o[c][vg] *= f;
            pT[m * 32 + mn]      = f2bf(p0);
            pT[m * 32 + 16 + mn] = f2bf(p1);
        }
        __syncthreads();

        // ---- O += P(16x32) x V(32x128); row sums via P x ones
        v16bf pf = ld_frag(&pT[mn * 32], half);
        osum = __builtin_amdgcn_wmma_f32_16x16x32_bf16(false, pf, false, ones, (short)0, osum, false, false);
        #pragma unroll
        for (int ch = 0; ch < 8; ++ch) {
            v16bf vb = ld_frag(&vTt[(ch * 16 + mn) * 32], half);
            o[ch] = __builtin_amdgcn_wmma_f32_16x16x32_bf16(false, pf, false, vb, (short)0, o[ch], false, false);
        }
        __syncthreads();
    }

    // ---- write partials
    const int comb = b * NKV_ + kv;
    const size_t base = ((size_t)comb * NSPLIT + split) * 16 * HD_;
    #pragma unroll
    for (int ch = 0; ch < 8; ++ch)
        #pragma unroll
        for (int vg = 0; vg < 8; ++vg) {
            int m = vg + 8 * half;
            acc_ws[base + (size_t)m * HD_ + ch * 16 + mn] = o[ch][vg];
        }
    if (mn == 0) {
        #pragma unroll
        for (int vg = 0; vg < 8; ++vg) {
            int m = vg + 8 * half;
            size_t mb = (((size_t)comb * NSPLIT + split) * 16 + m) * 2;
            ml_ws[mb]     = mi[vg];
            ml_ws[mb + 1] = osum[vg];
        }
    }
}

// ---------------------------------------------------------------------------
// Kernel 3: combine flash-decoding partials -> attn_ws (B,S,NH*HD)
// ---------------------------------------------------------------------------
__global__ __launch_bounds__(128)
void combine_kernel(const float* __restrict__ acc_ws,
                    const float* __restrict__ ml_ws,
                    float* __restrict__ attn_ws) {
    const int bid  = blockIdx.x;          // comb*16 + m
    const int comb = bid >> 4, m = bid & 15;
    const int b    = comb >> 3, kv = comb & 7;
    const int d    = threadIdx.x;

    float M = -1.0e30f;
    #pragma unroll
    for (int s = 0; s < NSPLIT; ++s)
        M = fmaxf(M, ml_ws[(((size_t)comb * NSPLIT + s) * 16 + m) * 2]);
    float L = 0.0f, O = 0.0f;
    #pragma unroll
    for (int s = 0; s < NSPLIT; ++s) {
        size_t mb = (((size_t)comb * NSPLIT + s) * 16 + m) * 2;
        float w = __expf(ml_ws[mb] - M);
        L += w * ml_ws[mb + 1];
        O += w * acc_ws[((size_t)comb * NSPLIT + s) * 16 * HD_ + (size_t)m * HD_ + d];
    }
    float res = O / L;
    int rep = m >> 2, spos = m & 3;
    int h = kv * REP_ + rep;
    attn_ws[((size_t)(b * S_ + spos) * NH_ + h) * HD_ + d] = res;
}

// ---------------------------------------------------------------------------
// Kernel 4: output projection attn(64x4096) @ wo(4096x4096) -> out
// ---------------------------------------------------------------------------
__global__ __launch_bounds__(32)
void out_proj_kernel(const float* __restrict__ attn_ws,
                     const float* __restrict__ wo,
                     float* __restrict__ out) {
    __shared__ __bf16 aT[16 * 32];
    __shared__ __bf16 wTt[16 * 32];
    const int l = threadIdx.x;
    const int half = l >> 4, mn = l & 15;
    const int m0 = blockIdx.x * 16, n0 = blockIdx.y * 16;

    v8f acc = {};
    #pragma unroll 1
    for (int k0 = 0; k0 < D_; k0 += 32) {
        #pragma unroll
        for (int q4 = 0; q4 < 4; ++q4) {
            int idx = l + 32 * q4;
            int r = idx >> 3, c4 = idx & 7;
            float4 va = *reinterpret_cast<const float4*>(attn_ws + (size_t)(m0 + r) * D_ + k0 + c4 * 4);
            st4bf(&aT[r * 32 + c4 * 4], va);
        }
        stage_w_tile(wTt, wo, D_, k0, n0, l);
        __syncthreads();
        v16bf a  = ld_frag(&aT[mn * 32], half);
        v16bf bf = ld_frag(&wTt[mn * 32], half);
        acc = __builtin_amdgcn_wmma_f32_16x16x32_bf16(false, a, false, bf, (short)0, acc, false, false);
        __syncthreads();
    }
    #pragma unroll
    for (int vg = 0; vg < 8; ++vg) {
        int m = m0 + vg + 8 * half;
        out[(size_t)m * D_ + n0 + mn] = acc[vg];
    }
}

// ---------------------------------------------------------------------------
extern "C" void kernel_launch(void* const* d_in, const int* in_sizes, int n_in,
                              void* d_out, int out_size, void* d_ws, size_t ws_size,
                              hipStream_t stream) {
    const float* x       = (const float*)d_in[0];
    const float* wq      = (const float*)d_in[1];
    const float* wk      = (const float*)d_in[2];
    const float* wv      = (const float*)d_in[3];
    const float* wo      = (const float*)d_in[4];
    const float* cache_k = (const float*)d_in[5];
    const float* cache_v = (const float*)d_in[6];
    const float* fcos    = (const float*)d_in[7];
    const float* fsin    = (const float*)d_in[8];
    // d_in[9] = mask (recomputed analytically), d_in[10] = start_pos (compile-time)

    float* ws      = (float*)d_ws;
    float* q_ws    = ws;
    float* k_ws    = q_ws  + (size_t)B_ * NH_  * S_ * HD_;
    float* v_ws    = k_ws  + (size_t)B_ * NKV_ * S_ * HD_;
    float* acc_ws  = v_ws  + (size_t)B_ * NKV_ * S_ * HD_;
    float* ml_ws   = acc_ws + (size_t)B_ * NKV_ * NSPLIT * 16 * HD_;
    float* attn_ws = ml_ws  + (size_t)B_ * NKV_ * NSPLIT * 16 * 2;

    qkv_rope_kernel<<<dim3(4, 384), 32, 0, stream>>>(x, wq, wk, wv, fcos, fsin,
                                                     q_ws, k_ws, v_ws);
    attn_kernel<<<dim3(NSPLIT, NKV_, B_), 32, 0, stream>>>(cache_k, cache_v,
                                                           q_ws, k_ws, v_ws,
                                                           acc_ws, ml_ws);
    combine_kernel<<<dim3(B_ * NKV_ * 16), 128, 0, stream>>>(acc_ws, ml_ws, attn_ws);
    out_proj_kernel<<<dim3(4, 256), 32, 0, stream>>>(attn_ws, wo, (float*)d_out);
}
